// DSTCEncoder_27015344292521
// MI455X (gfx1250) — matrix operand order repaired and checked by reading, not compiled, not run on hardware
//
#include <hip/hip_runtime.h>
#include <hip/hip_bf16.h>

typedef __attribute__((ext_vector_type(16))) _Float16 v16h;
typedef __attribute__((ext_vector_type(8)))  float    v8f;
typedef __attribute__((ext_vector_type(2)))  float    v2f;

union V16H { v16h v; uint4 q[2]; };
union V8F  { v8f  v; float f[8]; };

// ---------------------------------------------------------------------------
// Pack conv weights (KH,KW,Cin,Cout) f32 -> f16 transposed [Npad][Kpad],
// zero-padded in both dims so WMMA fragment loads are always in-bounds.
// ---------------------------------------------------------------------------
__global__ void pack_wt(const float* __restrict__ w, _Float16* __restrict__ wt,
                        int Cout, int K, int Kpad, int Npad) {
  int i = blockIdx.x * blockDim.x + threadIdx.x;
  int total = Npad * Kpad;
  if (i >= total) return;
  int n = i / Kpad;
  int k = i - n * Kpad;
  float v = 0.f;
  if (n < Cout && k < K) v = w[(size_t)k * Cout + n];   // w flat is k-major, Cout minor
  wt[(size_t)n * Kpad + k] = (_Float16)v;
}

// ---------------------------------------------------------------------------
// conv0: 3x3 stride-3 SAME(=no pad: 384 = 128*3), 3->16 ch, +bias, no act.
// One thread per (b,y,x); computes all 16 output channels; writes f32 + f16.
// ---------------------------------------------------------------------------
__global__ __launch_bounds__(256)
void conv0_kernel(const float* __restrict__ x, const float* __restrict__ w0,
                  const float* __restrict__ b0,
                  float* __restrict__ h32, _Float16* __restrict__ h16) {
  __shared__ float lw[432];   // 27*16
  __shared__ float lb[16];
  int t = threadIdx.x;
  for (int i = t; i < 432; i += 256) lw[i] = w0[i];
  if (t < 16) lb[t] = b0[t];
  __syncthreads();

  int m  = blockIdx.x * 256 + t;          // 0 .. 64*128*128-1
  int xo = m & 127;
  int yo = (m >> 7) & 127;
  int b  = m >> 14;

  float acc[16];
#pragma unroll
  for (int c = 0; c < 16; ++c) acc[c] = lb[c];

  const float* xb = x + (size_t)b * 384 * 384 * 3;
#pragma unroll
  for (int kh = 0; kh < 3; ++kh) {
    int iy = yo * 3 + kh;
#pragma unroll
    for (int kw = 0; kw < 3; ++kw) {
      int ix = xo * 3 + kw;
      const float* xp = xb + ((size_t)iy * 384 + ix) * 3;
      float x0 = xp[0], x1 = xp[1], x2 = xp[2];
      const float* wp = lw + (kh * 3 + kw) * 48;
#pragma unroll
      for (int c = 0; c < 16; ++c)
        acc[c] += x0 * wp[c] + x1 * wp[16 + c] + x2 * wp[32 + c];
    }
  }
  size_t ob = (size_t)m * 16;
#pragma unroll
  for (int c = 0; c < 16; ++c) {
    h32[ob + c] = acc[c];
    h16[ob + c] = (_Float16)acc[c];
  }
}

// ---------------------------------------------------------------------------
// Generic implicit-GEMM conv via v_wmma_f32_16x16x32_f16 (used for Cin==16
// layers where a 32-K chunk crosses tap boundaries). 4 waves/WG share a 16xK
// A tile staged through 1KB LDS chunks; wave w owns N-tile blockIdx.y*64+w*16.
// ---------------------------------------------------------------------------
__global__ __launch_bounds__(128)
void wmma_conv(const _Float16* __restrict__ Ain, const _Float16* __restrict__ Wt,
               const float* __restrict__ bias,
               float* __restrict__ out32, _Float16* __restrict__ out16,
               int H, int W, int lc,                // lc = log2(Cin)
               int sH, int sW, int padH, int padW,
               int lWO, int lHW,                    // log2(WO), log2(HO*WO)
               int Kdim, int Kpad, int N, int leaky) {
  __shared__ alignas(16) _Float16 lsA[16 * 32];

  int t    = threadIdx.x;
  int lane = t & 31;
  int hi   = lane >> 4;
  int row  = lane & 15;
  int wave = t >> 5;
  int m0   = blockIdx.x << 4;
  int n    = (blockIdx.y << 6) + (wave << 4) + row;
  int Cin  = 1 << lc;

  int srow = t >> 3;                 // 0..15
  int skk  = (t & 7) << 2;           // 0,4,...,28
  int sm   = m0 + srow;
  int sb   = sm >> lHW;
  int srem = sm & ((1 << lHW) - 1);
  int syo  = srem >> lWO;
  int sxo  = srem & ((1 << lWO) - 1);

  V8F c;
#pragma unroll
  for (int i = 0; i < 8; ++i) c.f[i] = 0.f;

  const _Float16* wrow = Wt + (size_t)n * Kpad;

  for (int k0 = 0; k0 < Kpad; k0 += 32) {
    int k = k0 + skk;
    unsigned long long val = 0ull;
    if (k < Kdim) {
      int ci   = k & (Cin - 1);
      int rest = k >> lc;
      int kh   = rest / 3;
      int kw   = rest - kh * 3;
      int iy   = syo * sH + kh - padH;
      int ix   = sxo * sW + kw - padW;
      if ((unsigned)iy < (unsigned)H && (unsigned)ix < (unsigned)W) {
        const _Float16* ap = Ain + ((((size_t)sb * H + iy) * W + ix) << lc) + ci;
        val = *(const unsigned long long*)ap;
      }
    }
    __syncthreads();
    *(unsigned long long*)&lsA[srow * 32 + skk] = val;
    __syncthreads();

    V16H a;
    const uint4* rowp = (const uint4*)&lsA[row * 32];
    a.q[0] = rowp[hi << 1];
    a.q[1] = rowp[(hi << 1) + 1];

    V16H bf;
    const uint4* wp = (const uint4*)(wrow + k0 + (hi << 4));
    bf.q[0] = wp[0];
    bf.q[1] = wp[1];

    c.v = __builtin_amdgcn_wmma_f32_16x16x32_f16(false, a.v, false, bf.v,
                                                 (short)0, c.v, false, false);
  }

  if (n < N) {
    float bv = bias ? bias[n] : 0.f;
#pragma unroll
    for (int i = 0; i < 8; ++i) {
      int   mr = m0 + i + (hi << 3);
      float v  = c.f[i] + bv;
      if (leaky) v = (v >= 0.f) ? v : 0.5f * v;
      size_t oi = (size_t)mr * N + n;
      if (out32) out32[oi] = v;
      if (out16) out16[oi] = (_Float16)v;
    }
  }
}

// ---------------------------------------------------------------------------
// Wide-channel implicit-GEMM conv (Cin in {64,128}: a 64-K chunk never crosses
// a tap boundary). Explicit kh/kw/ci0 loops: no division, incremental address
// math, 16B staging loads (b128) into a 2KB LDS tile, 2 WMMAs per sync pair.
// ---------------------------------------------------------------------------
__global__ __launch_bounds__(128)
void wmma_conv_big(const _Float16* __restrict__ Ain, const _Float16* __restrict__ Wt,
                   const float* __restrict__ bias,
                   float* __restrict__ out32, _Float16* __restrict__ out16,
                   int H, int W, int lc,            // Cin = 1<<lc, >= 64
                   int KH, int sH, int sW, int padH, int padW,
                   int lWO, int lHW,
                   int Kpad, int N, int leaky) {
  __shared__ alignas(16) _Float16 lsA[16 * 64];

  int t    = threadIdx.x;
  int lane = t & 31;
  int hi   = lane >> 4;
  int row  = lane & 15;
  int wave = t >> 5;
  int m0   = blockIdx.x << 4;
  int n    = (blockIdx.y << 6) + (wave << 4) + row;
  int Cin  = 1 << lc;

  int srow = t >> 3;                 // 0..15 : staged A row
  int s8   = (t & 7) << 3;           // halves 0,8,...,56 within 64-K chunk
  int sm   = m0 + srow;
  int sb   = sm >> lHW;
  int srem = sm & ((1 << lHW) - 1);
  int syo  = srem >> lWO;
  int sxo  = srem & ((1 << lWO) - 1);
  const _Float16* abase = Ain + (((size_t)sb * H * W) << lc);

  V8F c;
#pragma unroll
  for (int i = 0; i < 8; ++i) c.f[i] = 0.f;

  const _Float16* wrow = Wt + (size_t)n * Kpad;
  int kb = 0;

  for (int kh = 0; kh < KH; ++kh) {
    int iy = syo * sH + kh - padH;
#pragma unroll
    for (int kw = 0; kw < 3; ++kw) {
      int  ix  = sxo * sW + kw - padW;
      bool inb = ((unsigned)iy < (unsigned)H) && ((unsigned)ix < (unsigned)W);
      size_t spat = ((size_t)iy * W + ix);
      for (int ci0 = 0; ci0 < Cin; ci0 += 64, kb += 64) {
        uint4 val = make_uint4(0u, 0u, 0u, 0u);
        if (inb) val = *(const uint4*)(abase + (spat << lc) + ci0 + s8);
        __syncthreads();
        *(uint4*)&lsA[srow * 64 + s8] = val;
        __syncthreads();

        const uint4* rowp = (const uint4*)&lsA[row * 64];
        const uint4* wp   = (const uint4*)(wrow + kb);
#pragma unroll
        for (int j = 0; j < 2; ++j) {       // two 32-K WMMA steps
          V16H a, bf;
          a.q[0]  = rowp[j * 4 + (hi << 1)];     // K = j*32 + 16*hi   .. +7
          a.q[1]  = rowp[j * 4 + (hi << 1) + 1]; // K = j*32 + 16*hi+8 .. +7
          bf.q[0] = wp[j * 4 + (hi << 1)];  // K = j*32 + 16*hi   .. +15
          bf.q[1] = wp[j * 4 + (hi << 1) + 1];
          c.v = __builtin_amdgcn_wmma_f32_16x16x32_f16(false, a.v, false, bf.v,
                                                       (short)0, c.v, false, false);
        }
      }
    }
  }

  if (n < N) {
    float bv = bias ? bias[n] : 0.f;
#pragma unroll
    for (int i = 0; i < 8; ++i) {
      int   mr = m0 + i + (hi << 3);
      float v  = c.f[i] + bv;
      if (leaky) v = (v >= 0.f) ? v : 0.5f * v;
      size_t oi = (size_t)mr * N + n;
      if (out32) out32[oi] = v;
      if (out16) out16[oi] = (_Float16)v;
    }
  }
}

// ---------------------------------------------------------------------------
// Deformable resample: off (B,H,W,C,2) f32 -> bilinear sample of h (B,H,W,C)
// at clipped grid+offset, writing f16 for the next conv's A matrix.
// ---------------------------------------------------------------------------
__global__ __launch_bounds__(256)
void deform_bilinear(const float* __restrict__ hval, const float* __restrict__ off,
                     _Float16* __restrict__ dst,
                     int H, int W, int lC, int lW, int lH) {
  int i  = blockIdx.x * 256 + threadIdx.x;
  int C  = 1 << lC;
  int c  = i & (C - 1);
  int sp = i >> lC;                       // (b*H+y)*W + x
  int xg = sp & (W - 1);
  int yg = (sp >> lW) & (H - 1);
  int b  = sp >> (lW + lH);

  const float* op = off + (((size_t)sp) << (lC + 1)) + (c << 1);
  float dy = op[0], dx = op[1];
  float py = fminf(fmaxf((float)yg + dy, 0.f), (float)(H - 1));
  float px = fminf(fmaxf((float)xg + dx, 0.f), (float)(W - 1));
  float fy = floorf(py), fx = floorf(px);
  int y0 = (int)fy, x0 = (int)fx;
  int y1 = min(y0 + 1, H - 1), x1 = min(x0 + 1, W - 1);
  float wy = py - fy, wx = px - fx;

  auto idx = [&](int yy, int xx) -> size_t {
    return ((((size_t)b * H + yy) * W + xx) << lC) + c;
  };
  float v00 = hval[idx(y0, x0)], v01 = hval[idx(y0, x1)];
  float v10 = hval[idx(y1, x0)], v11 = hval[idx(y1, x1)];
  float top = v00 * (1.f - wx) + v01 * wx;
  float bot = v10 * (1.f - wx) + v11 * wx;
  dst[i] = (_Float16)(top * (1.f - wy) + bot * wy);
}

// ---------------------------------------------------------------------------
// Dense: (64,65536) @ (65536,512) + bd, fp32 WMMA 16x16x4 (streams wd once,
// no precision conversion of the 134MB weight matrix). One wave per 16x16
// output tile; grid (4,8) x 4 waves = 128 tiles.
// ---------------------------------------------------------------------------
__global__ __launch_bounds__(128)
void dense_wmma(const float* __restrict__ A, const float* __restrict__ Wd,
                const float* __restrict__ bd, float* __restrict__ out) {
  int t    = threadIdx.x;
  int lane = t & 31;
  int hi   = lane >> 4;
  int row  = lane & 15;
  int wave = t >> 5;
  int m0   = blockIdx.x << 4;
  int n    = ((blockIdx.y * 4 + wave) << 4) + row;

  const float* arow = A + (size_t)(m0 + row) * 65536;

  V8F c;
#pragma unroll
  for (int i = 0; i < 8; ++i) c.f[i] = 0.f;

#pragma unroll 8
  for (int k = 0; k < 65536; k += 4) {
    if ((k & 1023) == 0 && k + 2048 < 65536)
      __builtin_prefetch(&Wd[(size_t)(k + 2048) * 512 + n], 0, 1);
#if __has_builtin(__builtin_amdgcn_wmma_f32_16x16x4_f32)
    int ka = k + (hi << 1);
    v2f a, b;
    a[0] = arow[ka];
    a[1] = arow[ka + 1];
    b[0] = Wd[(size_t)ka * 512 + n];
    b[1] = Wd[(size_t)(ka + 1) * 512 + n];
    c.v = __builtin_amdgcn_wmma_f32_16x16x4_f32(false, a, false, b,
                                                (short)0, c.v, false, false);
#else
#pragma unroll
    for (int i = 0; i < 8; ++i) {
      int mr = m0 + i + (hi << 3);
      const float* ar = A + (size_t)mr * 65536 + k;
      float s = 0.f;
      for (int j = 0; j < 4; ++j) s += ar[j] * Wd[(size_t)(k + j) * 512 + n];
      c.f[i] += s;
    }
#endif
  }

  float bv = bd[n];
#pragma unroll
  for (int i = 0; i < 8; ++i) {
    int mr = m0 + i + (hi << 3);
    out[(size_t)mr * 512 + n] = c.f[i] + bv;
  }
}

// ---------------------------------------------------------------------------
static inline size_t align_up(size_t v, size_t a) { return (v + a - 1) / a * a; }

extern "C" void kernel_launch(void* const* d_in, const int* in_sizes, int n_in,
                              void* d_out, int out_size, void* d_ws, size_t ws_size,
                              hipStream_t stream) {
  const float* x   = (const float*)d_in[0];
  const float* w0  = (const float*)d_in[1];
  const float* b0  = (const float*)d_in[2];
  const float* wo1 = (const float*)d_in[3];
  const float* w1  = (const float*)d_in[4];
  const float* b1  = (const float*)d_in[5];
  const float* wo2 = (const float*)d_in[6];
  const float* w2  = (const float*)d_in[7];
  const float* b2  = (const float*)d_in[8];
  const float* wo3 = (const float*)d_in[9];
  const float* w3  = (const float*)d_in[10];
  const float* b3  = (const float*)d_in[11];
  const float* wd  = (const float*)d_in[12];
  const float* bd  = (const float*)d_in[13];
  float* out = (float*)d_out;

  char* base = (char*)d_ws;
  size_t off = 0;
  auto carve = [&](size_t bytes) -> void* {
    void* r = base + off;
    off = align_up(off + bytes, 256);
    return r;
  };
  const size_t ACT = (size_t)64 * 128 * 128 * 16;          // 16,777,216 elems
  float*    h32A  = (float*)carve(ACT * 4);                // h0 f32, then h2 f32
  float*    h32B  = (float*)carve(ACT * 4);                // h1 f32, then h3 f32
  _Float16* f16X  = (_Float16*)carve(ACT * 2);             // h0/h1/h2 f16
  _Float16* f16Y  = (_Float16*)carve(ACT * 2);             // deformed d1/d2/d3 f16
  float*    off32 = (float*)carve((size_t)64 * 128 * 128 * 32 * 4); // off1/2/3
  _Float16* wtO1  = (_Float16*)carve((size_t)64  * 160  * 2);
  _Float16* wtC1  = (_Float16*)carve((size_t)64  * 96   * 2);
  _Float16* wtO2  = (_Float16*)carve((size_t)128 * 576  * 2);
  _Float16* wtC2  = (_Float16*)carve((size_t)128 * 384  * 2);
  _Float16* wtO3  = (_Float16*)carve((size_t)256 * 1152 * 2);
  _Float16* wtC3  = (_Float16*)carve((size_t)256 * 768  * 2);

  auto pack = [&](const float* w, _Float16* wt, int Cout, int K, int Kpad, int Npad) {
    int total = Npad * Kpad;
    pack_wt<<<(total + 255) / 256, 256, 0, stream>>>(w, wt, Cout, K, Kpad, Npad);
  };
  pack(wo1, wtO1, 32, 144, 160, 64);
  pack(w1,  wtC1, 64, 96,  96,  64);
  pack(wo2, wtO2, 128, 576, 576, 128);
  pack(w2,  wtC2, 128, 384, 384, 128);
  pack(wo3, wtO3, 256, 1152, 1152, 256);
  pack(w3,  wtC3, 256, 768, 768, 256);

  // stage 0: conv0 3x3/s3, 3->16    (b,128,128,16)
  conv0_kernel<<<4096, 256, 0, stream>>>(x, w0, b0, h32A, f16X);

  // deform 1 @128x128, C=16 (Cin=16 -> generic kernel)
  wmma_conv<<<dim3(65536, 1), 128, 0, stream>>>(f16X, wtO1, nullptr, off32, nullptr,
      128, 128, 4, 1, 1, 1, 1, 7, 14, 144, 160, 32, 0);
  deform_bilinear<<<65536, 256, 0, stream>>>(h32A, off32, f16Y, 128, 128, 4, 7, 7);
  // conv1 (2,3)/s2, 16->64 -> (b,64,64,64), leaky (Cin=16 -> generic kernel)
  wmma_conv<<<dim3(16384, 1), 128, 0, stream>>>(f16Y, wtC1, b1, h32B, f16X,
      128, 128, 4, 2, 2, 0, 0, 6, 12, 96, 96, 64, 1);

  // deform 2 @64x64, C=64 (wide kernel)
  wmma_conv_big<<<dim3(16384, 2), 128, 0, stream>>>(f16X, wtO2, nullptr, off32, nullptr,
      64, 64, 6, 3, 1, 1, 1, 1, 6, 12, 576, 128, 0);
  deform_bilinear<<<65536, 256, 0, stream>>>(h32B, off32, f16Y, 64, 64, 6, 6, 6);
  // conv2 (2,3)/s2, 64->128 -> (b,32,32,128), leaky (wide kernel)
  wmma_conv_big<<<dim3(4096, 2), 128, 0, stream>>>(f16Y, wtC2, b2, h32A, f16X,
      64, 64, 6, 2, 2, 2, 0, 0, 5, 10, 384, 128, 1);

  // deform 3 @32x32, C=128 (wide kernel)
  wmma_conv_big<<<dim3(4096, 4), 128, 0, stream>>>(f16X, wtO3, nullptr, off32, nullptr,
      32, 32, 7, 3, 1, 1, 1, 1, 5, 10, 1152, 256, 0);
  deform_bilinear<<<32768, 256, 0, stream>>>(h32A, off32, f16Y, 32, 32, 7, 5, 5);
  // conv3 (2,3)/s2, 128->256 -> (b,16,16,256), leaky (wide kernel)
  wmma_conv_big<<<dim3(1024, 4), 128, 0, stream>>>(f16Y, wtC3, b3, h32B, nullptr,
      32, 32, 7, 2, 2, 2, 0, 0, 4, 8, 768, 256, 1);

  // dense (64,65536)@(65536,512)+bd
  dense_wmma<<<dim3(4, 8), 128, 0, stream>>>(h32B, wd, bd, out);

  (void)in_sizes; (void)n_in; (void)out_size; (void)ws_size;
}
